// GCNGraphClassifier_v2_57088705298676
// MI455X (gfx1250) — compile-verified, matrix operand order checked by reading, not run on hardware
//
#include <hip/hip_runtime.h>
#include <hip/hip_bf16.h>
#include <math.h>

#define N_NODES 50000
#define N_EDGES 800000
#define DIM     512
#define NB      64
#define NLAYER  4
#define NOUT    2
#define EPSV    1e-5f
#define KP      520   // padded LDS stride (bf16 elems) -> conflict-free frag loads

typedef __attribute__((ext_vector_type(16))) __bf16 v16bf;
typedef __attribute__((ext_vector_type(8)))  __bf16 v8bf;
typedef __attribute__((ext_vector_type(8)))  float  v8f;

// ---------------- graph normalization ----------------
__global__ void k_init_deg(float* deg) {
    int i = blockIdx.x * blockDim.x + threadIdx.x;
    if (i < N_NODES) deg[i] = 1.0f;                 // self-loop weight
}

__global__ void k_edge_deg(const long long* __restrict__ col,
                           const long long* __restrict__ same,
                           float* __restrict__ deg) {
    int e = blockIdx.x * blockDim.x + threadIdx.x;
    if (e < N_EDGES) {
        float w = (same[e] == 1) ? 1.0f : 0.7f;
        atomicAdd(&deg[(int)col[e]], w);
    }
}

__global__ void k_dis(float* __restrict__ deg, float* __restrict__ selfnorm) {
    int i = blockIdx.x * blockDim.x + threadIdx.x;
    if (i < N_NODES) {
        float d = rsqrtf(fmaxf(deg[i], EPSV));
        deg[i] = d;                                  // deg becomes dis in-place
        selfnorm[i] = d * d;
    }
}

__global__ void k_norm(const long long* __restrict__ row,
                       const long long* __restrict__ col,
                       const long long* __restrict__ same,
                       const float* __restrict__ dis,
                       float* __restrict__ norm) {
    int e = blockIdx.x * blockDim.x + threadIdx.x;
    if (e < N_EDGES) {
        float w = (same[e] == 1) ? 1.0f : 0.7f;
        norm[e] = dis[(int)row[e]] * w * dis[(int)col[e]];
    }
}

// ---------------- precision conversion ----------------
// Wc (L,K,N) f32 -> WcT (L,N,K) bf16, so B-fragments load contiguously
__global__ void k_cvt_w(const float* __restrict__ Wc, __bf16* __restrict__ wt) {
    int t = blockIdx.x * blockDim.x + threadIdx.x;
    if (t < NLAYER * DIM * DIM) {
        int l = t / (DIM * DIM);
        int r = t - l * DIM * DIM;
        int k = r / DIM, n = r - k * DIM;
        wt[(size_t)l * DIM * DIM + (size_t)n * DIM + k] = (__bf16)Wc[t];
    }
}

__global__ void k_cvt_x(const float* __restrict__ x, __bf16* __restrict__ hb) {
    int t = blockIdx.x * blockDim.x + threadIdx.x;
    if (t < N_NODES * DIM) hb[t] = (__bf16)x[t];
}

// ---------------- WMMA GEMM: hw = h_bf16 @ WcT^T  (N x 512 @ 512 x 512) ----------------
// block = 256 thr (8 waves). Block tile: 16 rows x 128 cols. Wave tile: 16x16.
__global__ __launch_bounds__(256)
void k_gemm(const __bf16* __restrict__ hb, const __bf16* __restrict__ wt,
            float* __restrict__ out) {
    __shared__ __bf16 As[16 * KP];
    const int m0  = blockIdx.y << 4;
    const int tid = threadIdx.x;

    // cooperative stage of 16x512 bf16 A-strip (64B per thread)
    {
        const int r = tid >> 4;
        const int k = (tid & 15) << 5;
        const uint4* src = (const uint4*)(hb + (size_t)(m0 + r) * DIM + k);
        uint4* dst = (uint4*)(&As[r * KP + k]);
        dst[0] = src[0]; dst[1] = src[1]; dst[2] = src[2]; dst[3] = src[3];
    }
    __syncthreads();

    const int wave = tid >> 5;
    const int lane = tid & 31;
    const int half = lane >> 4;      // K-half selector per ISA 16-bit layout
    const int l16  = lane & 15;
    const int n    = (blockIdx.x << 7) + (wave << 4) + l16;

    const __bf16* arow = &As[l16 * KP];
    const __bf16* brow = wt + (size_t)n * DIM;

    v8f acc = {};
    for (int kk = 0; kk < DIM; kk += 32) {
        v8bf alo = *(const v8bf*)(arow + kk + half * 8);
        v8bf ahi = *(const v8bf*)(arow + kk + 16 + half * 8);
        v8bf blo = *(const v8bf*)(brow + kk + half * 8);
        v8bf bhi = *(const v8bf*)(brow + kk + 16 + half * 8);
        v16bf a, b;
#pragma unroll
        for (int i = 0; i < 8; ++i) {
            a[i] = alo[i]; a[i + 8] = ahi[i];
            b[i] = blo[i]; b[i + 8] = bhi[i];
        }
        acc = __builtin_amdgcn_wmma_f32_16x16x32_bf16(
            /*neg_a=*/false, a, /*neg_b=*/false, b,
            /*c_mod=*/(short)0, acc, /*reuse_a=*/false, /*reuse_b=*/false);
    }

    // C layout: lanes 0-15 -> M=m0..m0+7, lanes 16-31 -> M=m0+8..m0+15, col = n
    const int rbase = m0 + half * 8;
#pragma unroll
    for (int j = 0; j < 8; ++j)
        out[(size_t)(rbase + j) * DIM + n] = acc[j];
}

// ---------------- aggregation ----------------
__global__ void k_agg_init(const float* __restrict__ hw,
                           const float* __restrict__ selfnorm,
                           const float* __restrict__ bcl,
                           float* __restrict__ agg) {
    int t = blockIdx.x * blockDim.x + threadIdx.x;  // over N*DIM/4
    if (t >= N_NODES * (DIM / 4)) return;
    int i = t / (DIM / 4);
    int c = (t - i * (DIM / 4)) << 2;
    float sn = selfnorm[i];
    float4 v = *(const float4*)(hw + (size_t)i * DIM + c);
    float4 bb = *(const float4*)(bcl + c);
    float4 o;
    o.x = bb.x + v.x * sn; o.y = bb.y + v.y * sn;
    o.z = bb.z + v.z * sn; o.w = bb.w + v.w * sn;
    *(float4*)(agg + (size_t)i * DIM + c) = o;
}

__global__ void k_agg_edges(const float* __restrict__ hw,
                            const long long* __restrict__ row,
                            const long long* __restrict__ col,
                            const float* __restrict__ norm,
                            float* __restrict__ agg) {
    int t = blockIdx.x * blockDim.x + threadIdx.x;  // over E*(DIM/4)
    if (t >= N_EDGES * (DIM / 4)) return;
    int e = t / (DIM / 4);
    int c = (t - e * (DIM / 4)) << 2;
    int r = (int)row[e], d = (int)col[e];
    float nv = norm[e];
    float4 v = *(const float4*)(hw + (size_t)r * DIM + c);
    float* dst = agg + (size_t)d * DIM + c;          // agg fits in 192MB L2
    atomicAdd(dst + 0, v.x * nv);
    atomicAdd(dst + 1, v.y * nv);
    atomicAdd(dst + 2, v.z * nv);
    atomicAdd(dst + 3, v.w * nv);
}

// ---------------- BN(inference) + ReLU; emit f32 (pooling) + bf16 (next GEMM) ----------------
__global__ void k_bn_relu(const float* __restrict__ agg,
                          const float* __restrict__ g, const float* __restrict__ bt,
                          const float* __restrict__ rm, const float* __restrict__ rv,
                          float* __restrict__ hf, __bf16* __restrict__ hb) {
    int t = blockIdx.x * blockDim.x + threadIdx.x;  // over N*DIM/4
    if (t >= N_NODES * (DIM / 4)) return;
    int i = t / (DIM / 4);
    int c = (t - i * (DIM / 4)) << 2;
    size_t idx = (size_t)i * DIM + c;
    float4 v  = *(const float4*)(agg + idx);
    float4 gg = *(const float4*)(g + c);
    float4 bb = *(const float4*)(bt + c);
    float4 mm = *(const float4*)(rm + c);
    float4 vv = *(const float4*)(rv + c);
    float y0 = fmaxf(gg.x * (v.x - mm.x) * rsqrtf(vv.x + EPSV) + bb.x, 0.0f);
    float y1 = fmaxf(gg.y * (v.y - mm.y) * rsqrtf(vv.y + EPSV) + bb.y, 0.0f);
    float y2 = fmaxf(gg.z * (v.z - mm.z) * rsqrtf(vv.z + EPSV) + bb.z, 0.0f);
    float y3 = fmaxf(gg.w * (v.w - mm.w) * rsqrtf(vv.w + EPSV) + bb.w, 0.0f);
    float4 o; o.x = y0; o.y = y1; o.z = y2; o.w = y3;
    *(float4*)(hf + idx) = o;
    hb[idx + 0] = (__bf16)y0; hb[idx + 1] = (__bf16)y1;
    hb[idx + 2] = (__bf16)y2; hb[idx + 3] = (__bf16)y3;
}

// ---------------- pooling ----------------
__global__ void k_pool_init(float* __restrict__ msum, float* __restrict__ mx,
                            float* __restrict__ cnt) {
    int t = blockIdx.x * blockDim.x + threadIdx.x;
    if (t < NB * DIM) { msum[t] = 0.0f; mx[t] = 0.0f; }
    if (t < NB) cnt[t] = 0.0f;
}

__global__ void k_count(const long long* __restrict__ batch, float* __restrict__ cnt) {
    int i = blockIdx.x * blockDim.x + threadIdx.x;
    if (i < N_NODES) atomicAdd(&cnt[(int)batch[i]], 1.0f);
}

__global__ void k_pool(const float* __restrict__ hf, const long long* __restrict__ batch,
                       float* __restrict__ msum, float* __restrict__ mx) {
    int t = blockIdx.x * blockDim.x + threadIdx.x;  // over N*DIM/4
    if (t >= N_NODES * (DIM / 4)) return;
    int i = t / (DIM / 4);
    int c = (t - i * (DIM / 4)) << 2;
    int b = (int)batch[i];
    float4 v = *(const float4*)(hf + (size_t)i * DIM + c);
    float* ms = msum + (size_t)b * DIM + c;
    unsigned* mu = (unsigned*)(mx + (size_t)b * DIM + c);
    atomicAdd(ms + 0, v.x); atomicMax(mu + 0, __float_as_uint(v.x));  // relu >= 0
    atomicAdd(ms + 1, v.y); atomicMax(mu + 1, __float_as_uint(v.y));
    atomicAdd(ms + 2, v.z); atomicMax(mu + 2, __float_as_uint(v.z));
    atomicAdd(ms + 3, v.w); atomicMax(mu + 3, __float_as_uint(v.w));
}

// ---------------- head: concat @ Wlin + log_softmax ----------------
__global__ void k_head(const float* __restrict__ msum, const float* __restrict__ mx,
                       const float* __restrict__ cnt, const float* __restrict__ Wlin,
                       const float* __restrict__ blin, float* __restrict__ out) {
    int b = threadIdx.x;
    if (b >= NB) return;
    float inv = 1.0f / fmaxf(cnt[b], 1.0f);
    float z0 = blin[0], z1 = blin[1];
    for (int c = 0; c < DIM; ++c) {
        float m = msum[(size_t)b * DIM + c] * inv;
        z0 += m * Wlin[c * NOUT + 0];
        z1 += m * Wlin[c * NOUT + 1];
    }
    for (int c = 0; c < DIM; ++c) {
        float m = mx[(size_t)b * DIM + c];
        z0 += m * Wlin[(DIM + c) * NOUT + 0];
        z1 += m * Wlin[(DIM + c) * NOUT + 1];
    }
    float mz  = fmaxf(z0, z1);
    float lse = mz + logf(expf(z0 - mz) + expf(z1 - mz));
    out[b * NOUT + 0] = z0 - lse;
    out[b * NOUT + 1] = z1 - lse;
}

// ---------------- launch ----------------
extern "C" void kernel_launch(void* const* d_in, const int* in_sizes, int n_in,
                              void* d_out, int out_size, void* d_ws, size_t ws_size,
                              hipStream_t stream) {
    (void)in_sizes; (void)n_in; (void)out_size; (void)ws_size;
    const float*     x     = (const float*)d_in[0];
    const long long* ei    = (const long long*)d_in[1];
    const long long* row   = ei;
    const long long* col   = ei + N_EDGES;
    const long long* batch = (const long long*)d_in[2];
    const long long* same  = (const long long*)d_in[3];
    const float*     Wc    = (const float*)d_in[4];
    const float*     bc    = (const float*)d_in[5];
    const float*     gamma = (const float*)d_in[6];
    const float*     beta  = (const float*)d_in[7];
    const float*     rmean = (const float*)d_in[8];
    const float*     rvar  = (const float*)d_in[9];
    const float*     Wlin  = (const float*)d_in[10];
    const float*     blin  = (const float*)d_in[11];
    float*           out   = (float*)d_out;

    // workspace carve-out (256B aligned)
    char* ws = (char*)d_ws;
    auto carve = [&](size_t bytes) {
        char* p = ws; ws += (bytes + 255) & ~(size_t)255; return p;
    };
    float*  dis  = (float*)carve((size_t)N_NODES * 4);          // deg -> dis in-place
    float*  sn   = (float*)carve((size_t)N_NODES * 4);
    float*  norm = (float*)carve((size_t)N_EDGES * 4);
    __bf16* wt   = (__bf16*)carve((size_t)NLAYER * DIM * DIM * 2);
    __bf16* hb   = (__bf16*)carve((size_t)N_NODES * DIM * 2);
    float*  hw   = (float*)carve((size_t)N_NODES * DIM * 4);
    float*  agg  = (float*)carve((size_t)N_NODES * DIM * 4);
    float*  msum = (float*)carve((size_t)NB * DIM * 4);
    float*  mx   = (float*)carve((size_t)NB * DIM * 4);
    float*  cnt  = (float*)carve((size_t)NB * 4);

    const int TB = 256;
    const int gN  = (N_NODES + TB - 1) / TB;                 // 196
    const int gE  = (N_EDGES + TB - 1) / TB;                 // 3125
    const int gND = (N_NODES * (DIM / 4) + TB - 1) / TB;     // 25000
    const int gED = (N_EDGES / TB) * (DIM / 4);              // 400000 (E divisible by TB)

    // graph normalization
    k_init_deg<<<gN, TB, 0, stream>>>(dis);
    k_edge_deg<<<gE, TB, 0, stream>>>(col, same, dis);
    k_dis<<<gN, TB, 0, stream>>>(dis, sn);
    k_norm<<<gE, TB, 0, stream>>>(row, col, same, dis, norm);

    // bf16 operand prep
    k_cvt_w<<<(NLAYER * DIM * DIM) / TB, TB, 0, stream>>>(Wc, wt);
    k_cvt_x<<<(N_NODES * DIM + TB - 1) / TB, TB, 0, stream>>>(x, hb);

    // GCN layers
    dim3 ggrid(DIM / 128, N_NODES / 16);                     // (4, 3125)
    for (int l = 0; l < NLAYER; ++l) {
        k_gemm<<<ggrid, TB, 0, stream>>>(hb, wt + (size_t)l * DIM * DIM, hw);
        k_agg_init<<<gND, TB, 0, stream>>>(hw, sn, bc + l * DIM, agg);
        k_agg_edges<<<gED, TB, 0, stream>>>(hw, row, col, norm, agg);
        k_bn_relu<<<gND, TB, 0, stream>>>(agg, gamma + l * DIM, beta + l * DIM,
                                          rmean + l * DIM, rvar + l * DIM, hw, hb);
    }

    // pooling + head
    k_pool_init<<<(NB * DIM + TB - 1) / TB, TB, 0, stream>>>(msum, mx, cnt);
    k_count<<<gN, TB, 0, stream>>>(batch, cnt);
    k_pool<<<gND, TB, 0, stream>>>(hw, batch, msum, mx);
    k_head<<<1, NB, 0, stream>>>(msum, mx, cnt, Wlin, blin, out);
}
